// HNMFocalLoss_38122129719835
// MI455X (gfx1250) — compile-verified, matrix operand order checked by reading, not run on hardware
//
#include <hip/hip_runtime.h>
#include <cstdint>

#define HN_B 32      // batch rows (reference constant)
#define NB1 4096     // level-0/1 histogram bins (12 bits)
#define NB3 256      // level-2 histogram bins (8 bits)
#define NBX 32       // streaming blocks per row
#define TPB 256      // threads per block (8 wave32)

// ---------------------------------------------------------------------------
// CDNA5 async global->LDS staging (ASYNCcnt-tracked), per-lane B128 copies.
// LDS byte offset == low 32 bits of the flat shared-aperture address.
// ---------------------------------------------------------------------------
__device__ __forceinline__ uint32_t lds_off(const void* p) {
  return (uint32_t)(uintptr_t)p;
}
__device__ __forceinline__ void async_copy16(uint32_t lds, const void* g) {
  asm volatile("global_load_async_to_lds_b128 %0, %1, off"
               :: "v"(lds), "v"(g) : "memory");
}
__device__ __forceinline__ void async_wait_le2() {
  asm volatile("s_wait_asynccnt 0x2" ::: "memory");
}
__device__ __forceinline__ void async_wait_le0() {
  asm volatile("s_wait_asynccnt 0x0" ::: "memory");
}

// Focal value; identical inlined code in every pass -> bit-identical recompute.
__device__ __forceinline__ float focal_val(float pr, float tg, bool& pos) {
  float p = fminf(fmaxf(pr, 1e-6f), 1.0f - 1e-6f);
  pos = (tg >= 0.5f);
  float pt = pos ? p : (1.0f - p);
  float at = pos ? 0.75f : 0.25f;
  float om = 1.0f - pt;
  float nl = -__logf(pt);
  return at * om * om * nl;   // >= 0 always -> u32 bits are order-monotone
}

// ---------------------------------------------------------------------------
// Histogram passes (LEVEL 0: bits[31:20] + n_pos; LEVEL 1: bits[19:8] within
// prefix; LEVEL 2: bits[7:0] within prefix). LDS-privatized, integer atomics.
// ---------------------------------------------------------------------------
template<int NB, int LEVEL>
__global__ __launch_bounds__(TPB)
void k_hist(const float* __restrict__ pred, const float* __restrict__ targ,
            uint32_t* __restrict__ g_hist, const uint32_t* __restrict__ g_state,
            uint32_t* __restrict__ g_npos, int N) {
  __shared__ uint32_t s_hist[NB];
  __shared__ uint32_t s_np, s_pre, s_flag;
  __shared__ float4 s_p[2][TPB];
  __shared__ float4 s_t[2][TPB];
  const int t = threadIdx.x;
  const int row = blockIdx.y;
  for (int i = t; i < NB; i += TPB) s_hist[i] = 0u;
  if (t == 0) {
    s_np = 0u;
    if (LEVEL > 0) { s_flag = g_state[row * 8 + 0]; s_pre = g_state[row * 8 + 3]; }
  }
  __syncthreads();
  if (LEVEL > 0 && s_flag) return;  // uniform early-out (no-positive fallback row)

  const int n4    = N >> 2;
  const int per   = n4 / (int)gridDim.x;
  const int iters = per / TPB;
  const float4* p4 = (const float4*)pred + (size_t)row * n4 + (size_t)blockIdx.x * per + t;
  const float4* t4 = (const float4*)targ + (size_t)row * n4 + (size_t)blockIdx.x * per + t;
  const uint32_t pre = (LEVEL > 0) ? s_pre : 0u;

  async_copy16(lds_off(&s_p[0][t]), p4);
  async_copy16(lds_off(&s_t[0][t]), t4);
  for (int j = 0; j < iters; ++j) {
    const int buf = j & 1;
    if (j + 1 < iters) {
      async_copy16(lds_off(&s_p[buf ^ 1][t]), p4 + (size_t)(j + 1) * TPB);
      async_copy16(lds_off(&s_t[buf ^ 1][t]), t4 + (size_t)(j + 1) * TPB);
      async_wait_le2();     // newest 2 may stay in flight; current tile ready
    } else {
      async_wait_le0();
    }
    float4 pv = s_p[buf][t];
    float4 tv = s_t[buf][t];
    float pp[4] = {pv.x, pv.y, pv.z, pv.w};
    float tt[4] = {tv.x, tv.y, tv.z, tv.w};
#pragma unroll
    for (int e = 0; e < 4; ++e) {
      bool pos;
      float f = focal_val(pp[e], tt[e], pos);
      uint32_t u = __float_as_uint(f);
      if (LEVEL == 0) {
        if (pos) atomicAdd(&s_np, 1u);
        else     atomicAdd(&s_hist[u >> 20], 1u);
      } else if (LEVEL == 1) {
        if (!pos && (u >> 20) == (pre >> 20)) atomicAdd(&s_hist[(u >> 8) & 0xFFFu], 1u);
      } else {
        if (!pos && (u >> 8) == (pre >> 8))   atomicAdd(&s_hist[u & 0xFFu], 1u);
      }
    }
  }
  __syncthreads();
  for (int i = t; i < NB; i += TPB) {
    uint32_t c = s_hist[i];
    if (c) atomicAdd(&g_hist[row * NB + i], c);
  }
  if (LEVEL == 0 && t == 0 && s_np) atomicAdd(&g_npos[row], s_np);
}

// ---------------------------------------------------------------------------
// Per-row radix-select step: find bin b with  cnt(>b) < k <= cnt(>=b).
// state[row*8]: 0 flag | 1 k | 2 npos | 3 prefixBits | 4 krem | 5 T | 6 k_at_T
// ---------------------------------------------------------------------------
__global__ __launch_bounds__(TPB)
void k_select(const uint32_t* __restrict__ g_hist, uint32_t* __restrict__ g_state,
              const uint32_t* __restrict__ g_npos, int level, int nbins, int N) {
  const int row = blockIdx.x;
  const int t = threadIdx.x;
  uint32_t* st = g_state + row * 8;
  __shared__ uint32_t sc[TPB];
  __shared__ uint32_t s_res[2];
  __shared__ uint32_t s_k;
  if (t == 0) {
    if (level == 0) {
      uint32_t np = g_npos[row];
      uint32_t nneg = (uint32_t)N - np;
      uint32_t k = np * 10u; if (k > nneg) k = nneg;
      st[1] = k; st[2] = np;
      if (np == 0u) { st[0] = 1u; st[5] = 0u; st[6] = 0u; s_k = 0u; }
      else          { st[0] = 0u; s_k = k; }
    } else {
      s_k = st[0] ? 0u : st[4];
    }
  }
  __syncthreads();
  const uint32_t k = s_k;
  if (k == 0u) return;  // uniform

  const uint32_t* hist = g_hist + row * nbins;
  const int chunk = nbins / TPB;  // 16 or 1
  uint32_t cs = 0u;
  for (int i = 0; i < chunk; ++i) cs += hist[t * chunk + i];
  sc[t] = cs;
  __syncthreads();
  for (int d = 1; d < TPB; d <<= 1) {       // Hillis-Steele inclusive scan
    uint32_t v = (t >= d) ? sc[t - d] : 0u;
    __syncthreads();
    sc[t] += v;
    __syncthreads();
  }
  const uint32_t total = sc[TPB - 1];
  const uint32_t suffix = total - sc[t];    // count in higher chunks
  if (suffix < k && k <= suffix + cs) {     // exactly one thread qualifies
    uint32_t acc = suffix;
    for (int i = chunk - 1; i >= 0; --i) {
      uint32_t c = hist[t * chunk + i];
      if (k <= acc + c) { s_res[0] = (uint32_t)(t * chunk + i); s_res[1] = acc; break; }
      acc += c;
    }
  }
  __syncthreads();
  if (t == 0) {
    const uint32_t b = s_res[0], above = s_res[1];
    if (level == 0)      { st[3] = b << 20;  st[4] = k - above; }
    else if (level == 1) { st[3] |= b << 8;  st[4] = k - above; }
    else                 { st[5] = st[3] | b; st[6] = k - above; }
  }
}

// ---------------------------------------------------------------------------
// Final streaming pass: deterministic block tree-reduce of pos_sum and
// sum(neg focal with bits > T).
// ---------------------------------------------------------------------------
__global__ __launch_bounds__(TPB)
void k_final(const float* __restrict__ pred, const float* __restrict__ targ,
             const uint32_t* __restrict__ g_state,
             float* __restrict__ Ppos, float* __restrict__ Pgt, int N) {
  __shared__ float4 s_p[2][TPB];
  __shared__ float4 s_t[2][TPB];
  __shared__ float s_a[TPB], s_b[TPB];
  __shared__ uint32_t s_T;
  const int t = threadIdx.x;
  const int row = blockIdx.y;
  if (t == 0) s_T = g_state[row * 8 + 5];
  __syncthreads();
  const uint32_t T = s_T;

  const int n4    = N >> 2;
  const int per   = n4 / (int)gridDim.x;
  const int iters = per / TPB;
  const float4* p4 = (const float4*)pred + (size_t)row * n4 + (size_t)blockIdx.x * per + t;
  const float4* t4 = (const float4*)targ + (size_t)row * n4 + (size_t)blockIdx.x * per + t;

  float fpos = 0.0f, fgt = 0.0f;
  async_copy16(lds_off(&s_p[0][t]), p4);
  async_copy16(lds_off(&s_t[0][t]), t4);
  for (int j = 0; j < iters; ++j) {
    const int buf = j & 1;
    if (j + 1 < iters) {
      async_copy16(lds_off(&s_p[buf ^ 1][t]), p4 + (size_t)(j + 1) * TPB);
      async_copy16(lds_off(&s_t[buf ^ 1][t]), t4 + (size_t)(j + 1) * TPB);
      async_wait_le2();
    } else {
      async_wait_le0();
    }
    float4 pv = s_p[buf][t];
    float4 tv = s_t[buf][t];
    float pp[4] = {pv.x, pv.y, pv.z, pv.w};
    float tt[4] = {tv.x, tv.y, tv.z, tv.w};
#pragma unroll
    for (int e = 0; e < 4; ++e) {
      bool pos;
      float f = focal_val(pp[e], tt[e], pos);
      uint32_t u = __float_as_uint(f);
      if (pos) fpos += f;
      else if (u > T) fgt += f;
    }
  }
  s_a[t] = fpos; s_b[t] = fgt;
  __syncthreads();
  for (int s = TPB >> 1; s > 0; s >>= 1) {  // fixed-shape deterministic reduce
    if (t < s) { s_a[t] += s_a[t + s]; s_b[t] += s_b[t + s]; }
    __syncthreads();
  }
  if (t == 0) {
    Ppos[row * (int)gridDim.x + blockIdx.x] = s_a[0];
    Pgt [row * (int)gridDim.x + blockIdx.x] = s_b[0];
  }
}

// ---------------------------------------------------------------------------
// Combine: reduce the 32x32 partial matrices with WMMA (rowsum = P x ones),
// then form per-row loss and the batch mean. Single wave32; EXEC all-ones.
// ---------------------------------------------------------------------------
typedef __attribute__((ext_vector_type(2)))  float    v2f;
typedef __attribute__((ext_vector_type(8)))  float    v8f;
typedef __attribute__((ext_vector_type(16))) _Float16 v16h;

#if defined(__has_builtin)
# if __has_builtin(__builtin_amdgcn_wmma_f32_16x16x4_f32)
#  define HAVE_WMMA_F32X4 1
# endif
#endif

__global__ __launch_bounds__(32)
void k_combine(const uint32_t* __restrict__ g_state,
               const float* __restrict__ Ppos, const float* __restrict__ Pgt,
               float* __restrict__ out, int N) {
  __shared__ float rsum[2][HN_B];
  const int lane = threadIdx.x;  // wave32
  for (int m = 0; m < 2; ++m) {
    const float* P = m ? Pgt : Ppos;
    for (int g = 0; g < 2; ++g) {
      const int row = g * 16 + (lane & 15);
      v8f acc = {0.f, 0.f, 0.f, 0.f, 0.f, 0.f, 0.f, 0.f};
#if defined(HAVE_WMMA_F32X4)
      const int koff = (lane < 16) ? 0 : 2;    // A 16x4 f32 layout: K = koff, koff+1
#pragma unroll
      for (int c = 0; c < NBX / 4; ++c) {
        v2f a; a.x = P[row * NBX + c * 4 + koff];
               a.y = P[row * NBX + c * 4 + koff + 1];
        v2f b; b.x = 1.0f; b.y = 1.0f;         // ones: layout-independent
        acc = __builtin_amdgcn_wmma_f32_16x16x4_f32(
            false, a, false, b, (short)0, acc, false, false);
      }
#else
      v16h a, b;                               // A 16x32 f16 layout
#pragma unroll
      for (int v = 0; v < 16; ++v) {
        int K = (v & 7) + ((v >> 3) << 4) + ((lane >> 4) << 3);
        a[v] = (_Float16)P[row * NBX + K];
        b[v] = (_Float16)1.0f;
      }
      acc = __builtin_amdgcn_wmma_f32_16x16x32_f16(
          false, a, false, b, (short)0, acc, false, false);
#endif
      // D layout: vgpr r holds M=r (lanes 0-15) and M=8+r (lanes 16-31)
      if (lane == 0) {
#pragma unroll
        for (int r = 0; r < 8; ++r) rsum[m][g * 16 + r] = acc[r];
      }
      if (lane == 16) {
#pragma unroll
        for (int r = 0; r < 8; ++r) rsum[m][g * 16 + 8 + r] = acc[r];
      }
    }
  }
  __syncthreads();
  if (lane == 0) {
    float s = 0.0f;
    for (int row = 0; row < HN_B; ++row) {
      const uint32_t* st = g_state + row * 8;
      float pos = rsum[0][row], gt = rsum[1][row];
      float per;
      if (st[0]) {
        per = gt / (float)N;  // no-positive fallback: mean over all negatives
      } else {
        float hard = gt + (float)st[6] * __uint_as_float(st[5]);  // ties at T
        uint32_t den = st[2] + st[1];
        if (den == 0u) den = 1u;
        per = (pos + hard) / (float)den;
      }
      s += per;
    }
    out[0] = s / (float)HN_B;
  }
}

// ---------------------------------------------------------------------------
extern "C" void kernel_launch(void* const* d_in, const int* in_sizes, int n_in,
                              void* d_out, int out_size, void* d_ws, size_t ws_size,
                              hipStream_t stream) {
  (void)n_in; (void)out_size; (void)ws_size;
  const float* pred = (const float*)d_in[0];
  const float* targ = (const float*)d_in[1];
  const int total = in_sizes[0];
  const int N = total / HN_B;  // 262144

  uint32_t* hist1 = (uint32_t*)d_ws;
  uint32_t* hist2 = hist1 + (size_t)HN_B * NB1;
  uint32_t* hist3 = hist2 + (size_t)HN_B * NB1;
  uint32_t* npos  = hist3 + (size_t)HN_B * NB3;
  uint32_t* state = npos + HN_B;
  float* Ppos = (float*)(state + (size_t)HN_B * 8);
  float* Pgt  = Ppos + (size_t)HN_B * NBX;
  const size_t zbytes = (size_t)((char*)(Pgt + (size_t)HN_B * NBX) - (char*)d_ws);
  hipMemsetAsync(d_ws, 0, zbytes, stream);  // ~1.1 MB, required every call

  dim3 sgrid(NBX, HN_B);
  k_hist<NB1, 0><<<sgrid, TPB, 0, stream>>>(pred, targ, hist1, state, npos, N);
  k_select<<<HN_B, TPB, 0, stream>>>(hist1, state, npos, 0, NB1, N);
  k_hist<NB1, 1><<<sgrid, TPB, 0, stream>>>(pred, targ, hist2, state, npos, N);
  k_select<<<HN_B, TPB, 0, stream>>>(hist2, state, npos, 1, NB1, N);
  k_hist<NB3, 2><<<sgrid, TPB, 0, stream>>>(pred, targ, hist3, state, npos, N);
  k_select<<<HN_B, TPB, 0, stream>>>(hist3, state, npos, 2, NB3, N);
  k_final<<<sgrid, TPB, 0, stream>>>(pred, targ, state, Ppos, Pgt, N);
  k_combine<<<1, 32, 0, stream>>>(state, Ppos, Pgt, (float*)d_out, N);
}
